// MambaBlock_41042707480728
// MI455X (gfx1250) — matrix operand order repaired and checked by reading, not compiled, not run on hardware
//
#include <hip/hip_runtime.h>
#include <hip/hip_bf16.h>

// ---------------------------------------------------------------------------
// Mamba block for MI455X (gfx1250): bf16 WMMA GEMMs with double-buffered
// async global->LDS staging (ASYNCcnt), VALU scan/conv.
// ---------------------------------------------------------------------------

typedef __attribute__((ext_vector_type(16))) __bf16 v16bf;
typedef __attribute__((ext_vector_type(8)))  float  v8f;

// builtin parameter types (from hipcc diagnostic: int __vector_size__(16/8) *)
typedef int v4i_ __attribute__((vector_size(16)));
typedef int v2i_ __attribute__((vector_size(8)));
#define AS1 __attribute__((address_space(1)))
#define AS3 __attribute__((address_space(3)))

#define BATCH    2
#define SEQ      1024
#define DMODEL   1024
#define DINNER   2048
#define DCONV    4
#define DTRANK   64
#define NSTATE   16
#define XDBL     96           // DTRANK + 2*NSTATE
#define MROWS    (BATCH*SEQ)  // 2048

__device__ __forceinline__ void async_ld_b128(void* l, const void* g) {
    __builtin_amdgcn_global_load_async_to_lds_b128(
        (AS1 v4i_*)(v4i_*)g, (AS3 v4i_*)(v4i_*)l, 0, 0);
}
__device__ __forceinline__ void async_ld_b64(void* l, const void* g) {
    __builtin_amdgcn_global_load_async_to_lds_b64(
        (AS1 v2i_*)(v2i_*)g, (AS3 v2i_*)(v2i_*)l, 0, 0);
}

// ---------------------------------------------------------------------------
// Elementwise f32 -> bf16 convert
// ---------------------------------------------------------------------------
__global__ __launch_bounds__(256) void cvt_f32_bf16(const float* __restrict__ s,
                                                    __bf16* __restrict__ d, int n) {
    int i = blockIdx.x * 256 + threadIdx.x;
    if (i < n) d[i] = (__bf16)s[i];
}

// Take first DTRANK columns of x_dbl (ld=96) -> bf16 [M, 64]
__global__ __launch_bounds__(256) void cvt_delta_cols(const float* __restrict__ xdbl,
                                                      __bf16* __restrict__ d) {
    int i = blockIdx.x * 256 + threadIdx.x;      // 0 .. MROWS*DTRANK-1
    int m = i >> 6, k = i & 63;
    d[i] = (__bf16)xdbl[(size_t)m * XDBL + k];
}

// ---------------------------------------------------------------------------
// bf16 WMMA GEMM:  C[M,N] = act( A[M,K] * Bw[N,K]^T + bias )
// A, Bw row-major bf16 (K contiguous).  block tile 32(M) x 256(N), K step 32.
// 8 waves: 2(M) x 4(N); each wave: 16x64 via 4 x v_wmma_f32_16x16x32_bf16.
// Slabs staged with GLOBAL_LOAD_ASYNC_TO_LDS_* into ping-pong LDS buffers.
// act: 0 = none, 1 = softplus
// ---------------------------------------------------------------------------
__global__ __launch_bounds__(256) void gemm_bf16_wmma(
    const __bf16* __restrict__ A, const __bf16* __restrict__ Bw,
    const float* __restrict__ bias, float* __restrict__ C,
    int M, int N, int K, int ldc, int act)
{
    __shared__ __bf16 lA[2][32 * 32];
    __shared__ __bf16 lB[2][256 * 32];

    const int tid  = threadIdx.x;
    const int lane = tid & 31;
    const int wid  = tid >> 5;
    const int wm   = wid >> 2;        // 0..1
    const int wn   = wid & 3;         // 0..3
    const int half = lane >> 4;       // half-wave select
    const int mrow = lane & 15;

    const int blockM = blockIdx.y * 32;
    const int blockN = blockIdx.x * 256;

    // per-thread staging coordinates (fixed across k-slabs)
    const int ar = tid >> 3;                 // A row 0..31
    const int ac = (tid & 7) * 4;            // A col group (4 bf16 = 8 bytes)

    // issue one slab's async copies: 1 x b64 (A) + 4 x b128 (B) per thread
    auto stage = [&](int buf, int k0) {
        async_ld_b64(&lA[buf][ar * 32 + ac],
                     A + (size_t)(blockM + ar) * K + k0 + ac);
        #pragma unroll
        for (int i = 0; i < 4; ++i) {
            int chunk = i * 256 + tid;
            int r   = chunk >> 2;
            int seg = (chunk & 3) * 8;       // 8 bf16 = 16 bytes
            int row = blockN + r; if (row >= N) row = N - 1;   // clamp (discarded tiles)
            async_ld_b128(&lB[buf][r * 32 + seg],
                          Bw + (size_t)row * K + k0 + seg);
        }
    };

    v8f acc[4] = {};
    const int nstages = K >> 5;

    stage(0, 0);
    for (int s = 0; s < nstages; ++s) {
        const int  cur  = s & 1;
        const bool more = (s + 1) < nstages;
        if (more) {
            stage(cur ^ 1, (s + 1) << 5);
            // in-order completion: <=5 outstanding => current buffer landed
            asm volatile("s_wait_asynccnt 0x5" ::: "memory");
        } else {
            asm volatile("s_wait_asynccnt 0x0" ::: "memory");
        }
        __syncthreads();

        const __bf16* lAb = lA[cur];
        const __bf16* lBb = lB[cur];

        // ---- gather A fragment per ISA 16-bit A layout (16x32, MxK) ----
        v16bf af;
        #pragma unroll
        for (int j = 0; j < 8; ++j) {
            int kb = (j < 4 ? 2 * j : 16 + 2 * (j - 4)) + 8 * half;
            const __bf16* p = &lAb[(wm * 16 + mrow) * 32 + kb];
            af[2 * j]     = p[0];
            af[2 * j + 1] = p[1];
        }
        // ---- gather B fragments (32x16, KxN) and issue 4 WMMAs ----
        #pragma unroll
        for (int t = 0; t < 4; ++t) {
            v16bf bfg;
            #pragma unroll
            for (int j = 0; j < 8; ++j) {
                int kb = (j < 4 ? 2 * j : 16 + 2 * (j - 4)) + 8 * half;
                const __bf16* p = &lBb[(wn * 64 + t * 16 + mrow) * 32 + kb];
                bfg[2 * j]     = p[0];
                bfg[2 * j + 1] = p[1];
            }
            acc[t] = __builtin_amdgcn_wmma_f32_16x16x32_bf16(
                false, af, false, bfg, (short)0, acc[t], false, false);
        }
        __syncthreads();   // all reads of `cur` done before it is refilled
    }

    // ---- epilogue: bias + activation, C/D layout (lane=N, vgpr r=M row) ----
    const int nb = blockN + wn * 64;
    #pragma unroll
    for (int t = 0; t < 4; ++t) {
        int col = nb + t * 16 + mrow;
        if (col < N) {
            float bv = bias ? bias[col] : 0.0f;
            #pragma unroll
            for (int r = 0; r < 8; ++r) {
                int row = blockM + wm * 16 + r + 8 * half;
                float v = acc[t][r] + bv;
                if (act == 1)                      // softplus
                    v = (v > 20.0f) ? v : logf(1.0f + __expf(v));
                if (row < M) C[(size_t)row * ldc + col] = v;
            }
        }
    }
}

// ---------------------------------------------------------------------------
// Depthwise causal conv (DCONV=4) + bias + SiLU.
// Input: x_b half of x_and_res, row layout [MROWS, 2*DINNER].
// Outputs: f32 (for scan u) and bf16 (for x_proj GEMM A).
// ---------------------------------------------------------------------------
__global__ __launch_bounds__(256) void conv_silu_kernel(
    const float* __restrict__ xres, const float* __restrict__ conv_w,
    const float* __restrict__ conv_b, float* __restrict__ uf,
    __bf16* __restrict__ ubf)
{
    int idx = blockIdx.x * 256 + threadIdx.x;     // over MROWS*DINNER
    int d = idx & (DINNER - 1);
    int ml = idx >> 11;                            // b*SEQ + l
    int l = ml & (SEQ - 1);
    float acc = conv_b[d];
    #pragma unroll
    for (int j = 0; j < DCONV; ++j) {
        int ls = l - (DCONV - 1) + j;
        if (ls >= 0)
            acc += conv_w[d * DCONV + j] *
                   xres[(size_t)(ml - l + ls) * (2 * DINNER) + d];
    }
    float s = acc / (1.0f + __expf(-acc));         // silu
    uf[idx]  = s;
    ubf[idx] = (__bf16)s;
}

// ---------------------------------------------------------------------------
// Selective scan: one thread per (b, d).  16 states in registers.
// Fused epilogue: y = (scan + u*D) * silu(res)  -> bf16
// ---------------------------------------------------------------------------
__global__ __launch_bounds__(256) void scan_kernel(
    const float* __restrict__ delta, const float* __restrict__ u,
    const float* __restrict__ xdbl,  const float* __restrict__ A_log,
    const float* __restrict__ Dp,    const float* __restrict__ xres,
    __bf16* __restrict__ ybf)
{
    const int tid = threadIdx.x;
    const int blk = blockIdx.x;                   // 16 blocks
    const int b   = blk >> 3;                     // 8 blocks per batch
    const int d   = ((blk & 7) << 8) + tid;       // 0..2047

    float a[NSTATE];
    #pragma unroll
    for (int n = 0; n < NSTATE; ++n)
        a[n] = -__expf(A_log[d * NSTATE + n]);
    float h[NSTATE] = {};
    const float Dd = Dp[d];

    __shared__ float sbc[32];                     // b[16] then c[16] per step

    for (int t = 0; t < SEQ; ++t) {
        const size_t row = (size_t)b * SEQ + t;
        if (tid < 32) sbc[tid] = xdbl[row * XDBL + DTRANK + tid];
        __syncthreads();

        const float dt = delta[row * DINNER + d];
        const float uu = u[row * DINNER + d];
        float y = 0.0f;
        #pragma unroll
        for (int n = 0; n < NSTATE; ++n) {
            float dA = __expf(dt * a[n]);
            h[n] = dA * h[n] + dt * sbc[n] * uu;
            y += h[n] * sbc[NSTATE + n];
        }
        const float res  = xres[row * (2 * DINNER) + DINNER + d];
        const float sres = res / (1.0f + __expf(-res));
        ybf[row * DINNER + d] = (__bf16)((y + uu * Dd) * sres);
        __syncthreads();
    }
}

// ---------------------------------------------------------------------------
// Host side
// ---------------------------------------------------------------------------
extern "C" void kernel_launch(void* const* d_in, const int* in_sizes, int n_in,
                              void* d_out, int out_size, void* d_ws, size_t ws_size,
                              hipStream_t stream) {
    const float* x          = (const float*)d_in[0];
    const float* in_proj_w  = (const float*)d_in[1];
    const float* in_proj_b  = (const float*)d_in[2];
    const float* conv_w     = (const float*)d_in[3];
    const float* conv_b     = (const float*)d_in[4];
    const float* x_proj_w   = (const float*)d_in[5];
    const float* dt_proj_w  = (const float*)d_in[6];
    const float* dt_proj_b  = (const float*)d_in[7];
    const float* A_log      = (const float*)d_in[8];
    const float* D_param    = (const float*)d_in[9];
    const float* out_proj_w = (const float*)d_in[10];
    const float* out_proj_b = (const float*)d_in[11];

    char* ws = (char*)d_ws;
    size_t off = 0;
    auto carve = [&](size_t bytes) { void* p = ws + off; off += (bytes + 255) & ~(size_t)255; return p; };

    __bf16* xbf      = (__bf16*)carve((size_t)MROWS * DMODEL * 2);        // 4 MB
    __bf16* w_in     = (__bf16*)carve((size_t)2 * DINNER * DMODEL * 2);   // 8 MB
    __bf16* w_xp     = (__bf16*)carve((size_t)XDBL * DINNER * 2);
    __bf16* w_dt     = (__bf16*)carve((size_t)DINNER * DTRANK * 2);
    __bf16* w_out    = (__bf16*)carve((size_t)DMODEL * DINNER * 2);       // 4 MB
    float*  xres     = (float*)carve((size_t)MROWS * 2 * DINNER * 4);     // 32 MB
    float*  uf       = (float*)carve((size_t)MROWS * DINNER * 4);         // 16 MB
    __bf16* ubf      = (__bf16*)carve((size_t)MROWS * DINNER * 2);        // 8 MB
    float*  x_dbl    = (float*)carve((size_t)MROWS * XDBL * 4);
    __bf16* din_bf   = (__bf16*)carve((size_t)MROWS * DTRANK * 2);
    float*  delta    = (float*)carve((size_t)MROWS * DINNER * 4);         // 16 MB
    __bf16* ybf      = (__bf16*)carve((size_t)MROWS * DINNER * 2);        // 8 MB
    (void)ws_size; (void)in_sizes; (void)n_in; (void)out_size;

    // 1) f32 -> bf16 conversions
    cvt_f32_bf16<<<(MROWS * DMODEL + 255) / 256, 256, 0, stream>>>(x, xbf, MROWS * DMODEL);
    cvt_f32_bf16<<<(2 * DINNER * DMODEL + 255) / 256, 256, 0, stream>>>(in_proj_w, w_in, 2 * DINNER * DMODEL);
    cvt_f32_bf16<<<(XDBL * DINNER + 255) / 256, 256, 0, stream>>>(x_proj_w, w_xp, XDBL * DINNER);
    cvt_f32_bf16<<<(DINNER * DTRANK + 255) / 256, 256, 0, stream>>>(dt_proj_w, w_dt, DINNER * DTRANK);
    cvt_f32_bf16<<<(DMODEL * DINNER + 255) / 256, 256, 0, stream>>>(out_proj_w, w_out, DMODEL * DINNER);

    // 2) in_proj: [2048,1024] x [4096,1024]^T -> [2048,4096]
    gemm_bf16_wmma<<<dim3((2 * DINNER + 255) / 256, MROWS / 32), 256, 0, stream>>>(
        xbf, w_in, in_proj_b, xres, MROWS, 2 * DINNER, DMODEL, 2 * DINNER, 0);

    // 3) depthwise causal conv + silu
    conv_silu_kernel<<<(MROWS * DINNER) / 256, 256, 0, stream>>>(xres, conv_w, conv_b, uf, ubf);

    // 4) x_proj: [2048,2048] x [96,2048]^T -> [2048,96]
    gemm_bf16_wmma<<<dim3(1, MROWS / 32), 256, 0, stream>>>(
        ubf, w_xp, nullptr, x_dbl, MROWS, XDBL, DINNER, XDBL, 0);

    // 5) delta-rank slice -> bf16
    cvt_delta_cols<<<(MROWS * DTRANK) / 256, 256, 0, stream>>>(x_dbl, din_bf);

    // 6) dt_proj + softplus: [2048,64] x [2048,64]^T -> [2048,2048]
    gemm_bf16_wmma<<<dim3(DINNER / 256, MROWS / 32), 256, 0, stream>>>(
        din_bf, w_dt, dt_proj_b, delta, MROWS, DINNER, DTRANK, DINNER, 1);

    // 7) selective scan + fused gating epilogue -> ybf
    scan_kernel<<<(BATCH * DINNER) / 256, 256, 0, stream>>>(
        delta, uf, x_dbl, A_log, D_param, xres, ybf);

    // 8) out_proj: [2048,2048] x [1024,2048]^T -> [2048,1024]
    gemm_bf16_wmma<<<dim3(DMODEL / 256, MROWS / 32), 256, 0, stream>>>(
        ybf, w_out, out_proj_b, (float*)d_out, MROWS, DMODEL, DINNER, DMODEL, 0);
}
